// GPT2_78692390798049
// MI455X (gfx1250) — matrix operand order repaired and checked by reading, not compiled, not run on hardware
//
#include <hip/hip_runtime.h>
#include <hip/hip_bf16.h>
#include <math.h>

// ---------------------------------------------------------------------------
// GPT-2 forward on MI455X (gfx1250, wave32, WMMA).
// All GEMMs run through v_wmma_f32_16x16x32_f16 with f32 accumulation.
// Block tile 128x128x32: 8 waves (2x4), each wave 64x32 = 4x2 WMMA accums
// -> 8 v_wmma per K-step. Next K-step's global loads are issued between the
// LDS-ready barrier and the WMMA work (software pipeline).
// Model dims: B=2, T=1024, E=1024, H=1, HS=1024, L=4, V=32000, HID=4096.
// Workspace requirement: 88 MB.
// ---------------------------------------------------------------------------

typedef __attribute__((ext_vector_type(16))) _Float16 v16h;
typedef __attribute__((ext_vector_type(8)))  _Float16 v8h;
typedef __attribute__((ext_vector_type(4)))  _Float16 v4h;
typedef __attribute__((ext_vector_type(8)))  float    v8f;
typedef __attribute__((ext_vector_type(4)))  float    v4f;

#define NB   2
#define SEQ  1024
#define EMB  1024
#define NL   4
#define VOC  32000
#define HID  4096
#define NROW (NB * SEQ)          // 2048

#define BM 128
#define BN 128
#define BK 32
#define LDSP 40                  // padded LDS row stride in halfs (80 B, 16B-aligned)

// ---------------------------------------------------------------------------
// Generic WMMA GEMM: C[b] = A[b] (f32,[M,K]) * B[b] (f32) [+bias] [+residual]
//   bT==0 : B stored row-major [K,N]   (weight GEMMs, a@v)
//   bT==1 : B stored row-major [N,K]   (q @ k^T)
// ---------------------------------------------------------------------------
__global__ __launch_bounds__(256) void wmma_gemm(
    const float* __restrict__ A, long long sA, int lda,
    const float* __restrict__ B, long long sB, int ldb, int bT,
    float*       __restrict__ C, long long sC, int ldc,
    const float* __restrict__ bias,
    const float* __restrict__ R, long long sR,
    int K)
{
  __shared__ _Float16 As[BM * LDSP];   // [M][K] tile, f16
  __shared__ _Float16 Bs[BN * LDSP];   // [N][K] tile, f16

  const int tid  = threadIdx.x;
  const int lane = tid & 31;
  const int wid  = tid >> 5;   // 0..7
  const int wM   = wid >> 2;   // 0..1  (64 rows each)
  const int wN   = wid & 3;    // 0..3  (32 cols each)
  const int g    = lane >> 4;  // lane group 0/1
  const int l16  = lane & 15;

  const int m0 = blockIdx.y * BM;
  const int n0 = blockIdx.x * BN;
  const int bz = blockIdx.z;

  const float* Ab = A + (long long)bz * sA;
  const float* Bb = B + (long long)bz * sB;
  float*       Cb = C + (long long)bz * sC;
  const float* Rb = R ? (R + (long long)bz * sR) : nullptr;

  // staging-thread coordinates
  const int c4a = (tid & 7) * 4;   // float4 column within 32-wide K (A / bT-B)
  const int rb  = tid >> 3;        // 0..31 row base (A / bT-B)
  const int kb4 = (tid >> 5) * 4;  // K base 0..28 (NN-B)
  const int nB4 = (tid & 31) * 4;  // N base 0..124 (NN-B)

  v8f acc[4][2];
#pragma unroll
  for (int mi = 0; mi < 4; ++mi)
#pragma unroll
    for (int ni = 0; ni < 2; ++ni)
#pragma unroll
      for (int e = 0; e < 8; ++e) acc[mi][ni][e] = 0.0f;

  v4f aReg[4], bReg[4];

  auto loadA = [&](int kk) {
#pragma unroll
    for (int p = 0; p < 4; ++p) {
      const int r = rb + p * 32;
      aReg[p] = *(const v4f*)(Ab + (long long)(m0 + r) * lda + (kk + c4a));
    }
  };
  auto loadB = [&](int kk) {
    if (bT) {
#pragma unroll
      for (int p = 0; p < 4; ++p) {
        const int n = rb + p * 32;
        bReg[p] = *(const v4f*)(Bb + (long long)(n0 + n) * ldb + (kk + c4a));
      }
    } else {
#pragma unroll
      for (int j = 0; j < 4; ++j)
        bReg[j] = *(const v4f*)(Bb + (long long)(kk + kb4 + j) * ldb + (n0 + nB4));
    }
  };

  loadA(0);
  loadB(0);

  for (int k0 = 0; k0 < K; k0 += BK) {
    // ---- convert staged registers to f16 and store to LDS ----
#pragma unroll
    for (int p = 0; p < 4; ++p) {
      const int r = rb + p * 32;
      v4h h;
#pragma unroll
      for (int i = 0; i < 4; ++i) h[i] = (_Float16)aReg[p][i];
      *(v4h*)(&As[r * LDSP + c4a]) = h;
    }
    if (bT) {
#pragma unroll
      for (int p = 0; p < 4; ++p) {
        const int n = rb + p * 32;
        v4h h;
#pragma unroll
        for (int i = 0; i < 4; ++i) h[i] = (_Float16)bReg[p][i];
        *(v4h*)(&Bs[n * LDSP + c4a]) = h;
      }
    } else {
      // bReg[j][jn] = B[k0+kb4+j][n0+nB4+jn]; transpose 4x4 in registers
#pragma unroll
      for (int jn = 0; jn < 4; ++jn) {
        v4h h;
#pragma unroll
        for (int j = 0; j < 4; ++j) h[j] = (_Float16)bReg[j][jn];
        *(v4h*)(&Bs[(nB4 + jn) * LDSP + kb4]) = h;
      }
    }
    __syncthreads();

    // ---- issue next tile's global loads (latency hidden under WMMA) ----
    const int kn = k0 + BK;
    if (kn < K) { loadA(kn); loadB(kn); }

    // ---- fragments per ISA 16-bit layouts ----
    // A (16x32, MxK): lane holds M=l16; halfs K=[g*8,+8) and [16+g*8,+8)
    // B (32x16, KxN): lane holds N=l16; halfs K=[g*8,+8) and [16+g*8,+8)
    v16h af[4], bf[2];
#pragma unroll
    for (int mi = 0; mi < 4; ++mi) {
      const _Float16* ap = &As[(wM * 64 + mi * 16 + l16) * LDSP];
      const v8h lo = *(const v8h*)(ap + g * 8);
      const v8h hi = *(const v8h*)(ap + 16 + g * 8);
#pragma unroll
      for (int i = 0; i < 8; ++i) { af[mi][i] = lo[i]; af[mi][8 + i] = hi[i]; }
    }
#pragma unroll
    for (int ni = 0; ni < 2; ++ni) {
      const _Float16* bp = &Bs[(wN * 32 + ni * 16 + l16) * LDSP];
      const v8h lo = *(const v8h*)(bp + g * 8);
      const v8h hi = *(const v8h*)(bp + 16 + g * 8);
#pragma unroll
      for (int i = 0; i < 8; ++i) { bf[ni][i] = lo[i]; bf[ni][8 + i] = hi[i]; }
    }

#pragma unroll
    for (int mi = 0; mi < 4; ++mi)
#pragma unroll
      for (int ni = 0; ni < 2; ++ni)
        acc[mi][ni] = __builtin_amdgcn_wmma_f32_16x16x32_f16(
            false, af[mi], false, bf[ni], (short)0, acc[mi][ni], false, false);
    __syncthreads();
  }

  // ---- epilogue: D layout VGPR e -> row = g*8+e, col = l16 ----
#pragma unroll
  for (int mi = 0; mi < 4; ++mi)
#pragma unroll
    for (int ni = 0; ni < 2; ++ni) {
      const int col = n0 + wN * 32 + ni * 16 + l16;
      const float bv = bias ? bias[col] : 0.0f;
#pragma unroll
      for (int e = 0; e < 8; ++e) {
        const int row = m0 + wM * 64 + mi * 16 + g * 8 + e;
        float val = acc[mi][ni][e] + bv;
        if (Rb) val += Rb[(long long)row * ldc + col];
        Cb[(long long)row * ldc + col] = val;
      }
    }
}

// ---------------------------------------------------------------------------
// Embedding: x[row] = vocab_emb[tokens[row]] + pos_emb[row % T]
// ---------------------------------------------------------------------------
__global__ __launch_bounds__(256) void embed_kernel(
    const int* __restrict__ tokens, const float* __restrict__ ve,
    const float* __restrict__ pe, float* __restrict__ x)
{
  const int row = blockIdx.x;
  const int t = row & (SEQ - 1);
  const int tok = tokens[row];
  const float* v = ve + (long long)tok * EMB;
  const float* p = pe + (long long)t * EMB;
  float* y = x + (long long)row * EMB;
  for (int i = threadIdx.x; i < EMB; i += 256) y[i] = v[i] + p[i];
}

// ---------------------------------------------------------------------------
// LayerNorm over last dim C (biased variance), optional exact GELU.
// One block per row; in==out allowed.
// ---------------------------------------------------------------------------
__global__ __launch_bounds__(256) void ln_kernel(
    const float* __restrict__ in, float* __restrict__ out,
    const float* __restrict__ gam, const float* __restrict__ bet,
    int C, int doGelu)
{
  const int row = blockIdx.x;
  const float* x = in + (long long)row * C;
  float* y = out + (long long)row * C;
  __shared__ float red[256];
  const int tid = threadIdx.x;

  float s = 0.0f;
  for (int i = tid; i < C; i += 256) s += x[i];
  red[tid] = s; __syncthreads();
  for (int st = 128; st > 0; st >>= 1) {
    if (tid < st) red[tid] += red[tid + st];
    __syncthreads();
  }
  const float mean = red[0] / (float)C;
  __syncthreads();

  float s2 = 0.0f;
  for (int i = tid; i < C; i += 256) { const float d = x[i] - mean; s2 += d * d; }
  red[tid] = s2; __syncthreads();
  for (int st = 128; st > 0; st >>= 1) {
    if (tid < st) red[tid] += red[tid + st];
    __syncthreads();
  }
  const float inv = rsqrtf(red[0] / (float)C + 1e-5f);
  __syncthreads();

  for (int i = tid; i < C; i += 256) {
    float v = (x[i] - mean) * inv * gam[i] + bet[i];
    if (doGelu) v = 0.5f * v * (1.0f + erff(v * 0.70710678118654752f));
    y[i] = v;
  }
}

// ---------------------------------------------------------------------------
// Causal softmax over rows of s [B*T, T], in place; applies SCALE first.
// ---------------------------------------------------------------------------
__global__ __launch_bounds__(256) void softmax_kernel(float* __restrict__ s, float scale)
{
  const int row = blockIdx.x;           // b*T + t
  const int t = row & (SEQ - 1);
  const int n = t + 1;                  // valid (unmasked) length
  float* p = s + (long long)row * SEQ;
  __shared__ float red[256];
  const int tid = threadIdx.x;

  float mx = -3.0e38f;
  for (int i = tid; i < n; i += 256) mx = fmaxf(mx, p[i] * scale);
  red[tid] = mx; __syncthreads();
  for (int st = 128; st > 0; st >>= 1) {
    if (tid < st) red[tid] = fmaxf(red[tid], red[tid + st]);
    __syncthreads();
  }
  const float m = red[0];
  __syncthreads();

  float sum = 0.0f;
  for (int i = tid; i < n; i += 256) sum += __expf(p[i] * scale - m);
  red[tid] = sum; __syncthreads();
  for (int st = 128; st > 0; st >>= 1) {
    if (tid < st) red[tid] += red[tid + st];
    __syncthreads();
  }
  const float inv = 1.0f / red[0];
  __syncthreads();

  for (int i = tid; i < SEQ; i += 256)
    p[i] = (i < n) ? __expf(p[i] * scale - m) * inv : 0.0f;
}

// ---------------------------------------------------------------------------
// Host orchestration
// ---------------------------------------------------------------------------
extern "C" void kernel_launch(void* const* d_in, const int* in_sizes, int n_in,
                              void* d_out, int out_size, void* d_ws, size_t ws_size,
                              hipStream_t stream) {
  (void)in_sizes; (void)n_in; (void)out_size; (void)ws_size;

  const int*   tokens = (const int*)  d_in[0];
  const float* vocab  = (const float*)d_in[1];
  const float* pos    = (const float*)d_in[2];
  const float* ln1_g  = (const float*)d_in[3];
  const float* ln1_b  = (const float*)d_in[4];
  const float* Wq     = (const float*)d_in[5];
  const float* Wk     = (const float*)d_in[6];
  const float* Wv     = (const float*)d_in[7];
  const float* Wo     = (const float*)d_in[8];
  const float* bo     = (const float*)d_in[9];
  const float* ln2_g  = (const float*)d_in[10];
  const float* ln2_b  = (const float*)d_in[11];
  const float* W1     = (const float*)d_in[12];
  const float* b1     = (const float*)d_in[13];
  const float* lnm_g  = (const float*)d_in[14];
  const float* lnm_b  = (const float*)d_in[15];
  const float* W2     = (const float*)d_in[16];
  const float* b2     = (const float*)d_in[17];
  const float* lnf_g  = (const float*)d_in[18];
  const float* lnf_b  = (const float*)d_in[19];
  const float* Wlm    = (const float*)d_in[20];
  const float* blm    = (const float*)d_in[21];
  float* logits = (float*)d_out;

  // Workspace layout (88 MB)
  char* ws = (char*)d_ws;
  const size_t MB = 1u << 20;
  float* x   = (float*)(ws + 0 * MB);   // [2048,1024]   8 MB
  float* h   = (float*)(ws + 8 * MB);   // [2048,1024]   8 MB
  float* q   = (float*)(ws + 16 * MB);  // [2048,1024]   8 MB
  float* kx  = (float*)(ws + 24 * MB);  // [2048,1024]   8 MB
  float* vx  = (float*)(ws + 32 * MB);  // [2048,1024]   8 MB
  float* sc  = (float*)(ws + 40 * MB);  // [2,1024,1024] 8 MB
  float* ob  = (float*)(ws + 48 * MB);  // [2048,1024]   8 MB
  float* mid = (float*)(ws + 56 * MB);  // [2048,4096]  32 MB

  const float SCALE = 0.03125f;  // (H*HS)^-0.5 = 1024^-0.5
  const long long BT2 = (long long)SEQ * EMB;   // per-batch activation stride
  const long long TT  = (long long)SEQ * SEQ;

  const dim3 blk(256);

  // x = vocab_emb[tokens] + pos_emb
  embed_kernel<<<NROW, blk, 0, stream>>>(tokens, vocab, pos, x);

  for (int l = 0; l < NL; ++l) {
    const float* wq  = Wq + (long long)l * EMB * EMB;
    const float* wk  = Wk + (long long)l * EMB * EMB;
    const float* wv  = Wv + (long long)l * EMB * EMB;
    const float* wo  = Wo + (long long)l * EMB * EMB;
    const float* bol = bo + (long long)l * EMB;
    const float* w1  = W1 + (long long)l * EMB * HID;
    const float* b1l = b1 + (long long)l * HID;
    const float* w2  = W2 + (long long)l * HID * EMB;
    const float* b2l = b2 + (long long)l * EMB;

    // h = LN1(x)
    ln_kernel<<<NROW, blk, 0, stream>>>(x, h, ln1_g + (long long)l * EMB,
                                        ln1_b + (long long)l * EMB, EMB, 0);

    // q,k,v = h @ W{q,k,v}   [2048,1024] x [1024,1024]
    {
      dim3 grid(EMB / BN, NROW / BM, 1);
      wmma_gemm<<<grid, blk, 0, stream>>>(h, 0, EMB, wq, 0, EMB, 0, q, 0, EMB,
                                          nullptr, nullptr, 0, EMB);
      wmma_gemm<<<grid, blk, 0, stream>>>(h, 0, EMB, wk, 0, EMB, 0, kx, 0, EMB,
                                          nullptr, nullptr, 0, EMB);
      wmma_gemm<<<grid, blk, 0, stream>>>(h, 0, EMB, wv, 0, EMB, 0, vx, 0, EMB,
                                          nullptr, nullptr, 0, EMB);
    }

    // s = q @ k^T  per batch   [1024,1024] x [1024,1024]^T
    {
      dim3 grid(SEQ / BN, SEQ / BM, NB);
      wmma_gemm<<<grid, blk, 0, stream>>>(q, BT2, EMB, kx, BT2, EMB, 1,
                                          sc, TT, SEQ, nullptr, nullptr, 0, EMB);
    }

    // causal softmax(s * SCALE)
    softmax_kernel<<<NROW, blk, 0, stream>>>(sc, SCALE);

    // o = a @ v  per batch
    {
      dim3 grid(EMB / BN, SEQ / BM, NB);
      wmma_gemm<<<grid, blk, 0, stream>>>(sc, TT, SEQ, vx, BT2, EMB, 0,
                                          ob, BT2, EMB, nullptr, nullptr, 0, SEQ);
    }

    // x = x + o @ Wo + bo
    {
      dim3 grid(EMB / BN, NROW / BM, 1);
      wmma_gemm<<<grid, blk, 0, stream>>>(ob, 0, EMB, wo, 0, EMB, 0, x, 0, EMB,
                                          bol, x, 0, EMB);
    }

    // h = LN2(x)
    ln_kernel<<<NROW, blk, 0, stream>>>(x, h, ln2_g + (long long)l * EMB,
                                        ln2_b + (long long)l * EMB, EMB, 0);

    // mid = h @ W1 + b1      [2048,1024] x [1024,4096]
    {
      dim3 grid(HID / BN, NROW / BM, 1);
      wmma_gemm<<<grid, blk, 0, stream>>>(h, 0, EMB, w1, 0, HID, 0, mid, 0, HID,
                                          b1l, nullptr, 0, EMB);
    }

    // mid = gelu(LN_m(mid))  in place
    ln_kernel<<<NROW, blk, 0, stream>>>(mid, mid, lnm_g + (long long)l * HID,
                                        lnm_b + (long long)l * HID, HID, 1);

    // x = x + mid @ W2 + b2  [2048,4096] x [4096,1024]
    {
      dim3 grid(EMB / BN, NROW / BM, 1);
      wmma_gemm<<<grid, blk, 0, stream>>>(mid, 0, HID, w2, 0, EMB, 0, x, 0, EMB,
                                          b2l, x, 0, HID);
    }
  }

  // h = LN_f(x);  logits = h @ Wlm + blm   [2048,1024] x [1024,32000]
  ln_kernel<<<NROW, blk, 0, stream>>>(x, h, lnf_g, lnf_b, EMB, 0);
  {
    dim3 grid(VOC / BN, NROW / BM, 1);
    wmma_gemm<<<grid, blk, 0, stream>>>(h, 0, EMB, Wlm, 0, VOC, 0, logits, 0, VOC,
                                        blm, nullptr, 0, EMB);
  }
}
